// SSA_max_24283745092271
// MI455X (gfx1250) — 16-bit and 8-bit matrix operand order repaired and checked by reading, compiled, not run on hardware
//
#include <hip/hip_runtime.h>
#include <hip/hip_bf16.h>
#include <stdint.h>

#define T_ 4
#define B_ 8
#define N_ 512
#define D_ 512
#define H_ 8
#define DH_ 64
#define TBN_ (T_ * B_ * N_)
#define INV_TAU 0.5f
#define VTH_ 1.0f
#define SCALE_ 0.044194173824159216f

typedef __attribute__((ext_vector_type(16))) __bf16 v16bf;
typedef __attribute__((ext_vector_type(8)))  float  v8f;
typedef __attribute__((ext_vector_type(8)))  int    v8i;

__device__ __forceinline__ v8f wmma_bf16(v16bf a, v16bf b, v8f c) {
  return __builtin_amdgcn_wmma_f32_16x16x32_bf16(false, a, false, b, (short)0, c,
                                                 false, false);
}
__device__ __forceinline__ v8i wmma_iu8(v8i a, v8i b, v8i c) {
  return __builtin_amdgcn_wmma_i32_16x16x64_iu8(false, a, false, b, c, false, false);
}

__device__ __forceinline__ void async_ld_b128(void* lds_dst, const void* gsrc) {
  uint32_t l = (uint32_t)(uintptr_t)lds_dst;
  uint64_t g = (uint64_t)(uintptr_t)gsrc;
  asm volatile("global_load_async_to_lds_b128 %0, %1, off"
               :: "v"(l), "v"(g) : "memory");
}
__device__ __forceinline__ void wait_async0() {
  asm volatile("s_wait_asynccnt 0" ::: "memory");
}

__device__ __forceinline__ v16bf ldsA_bf16(const __bf16* lds, int rowbase, int ld, int lane) {
  const __bf16* p = lds + (rowbase + (lane & 15)) * ld + ((lane >> 4) << 3);
  union { v16bf v; uint4 q[2]; } u;
  u.q[0] = *(const uint4*)(p);
  u.q[1] = *(const uint4*)(p + 16);
  return u.v;
}
__device__ __forceinline__ v16bf ldsA_u8bf(const uint8_t* lds, int rowbase, int ld, int lane) {
  const uint8_t* p = lds + (rowbase + (lane & 15)) * ld + ((lane >> 4) << 3);
  union { uint2 d; uint8_t b[8]; } lo, hi;
  lo.d = *(const uint2*)(p);
  hi.d = *(const uint2*)(p + 16);
  v16bf r;
#pragma unroll
  for (int i = 0; i < 8; ++i) {
    r[i] = (__bf16)(float)lo.b[i];
    r[i + 8] = (__bf16)(float)hi.b[i];
  }
  return r;
}
__device__ __forceinline__ v16bf ldsB_bf16(const __bf16* ldsT, int colbase, int ld, int lane) {
  const __bf16* p = ldsT + (colbase + (lane & 15)) * ld + ((lane >> 4) << 3);
  union { v16bf v; uint4 q[2]; } u;
  u.q[0] = *(const uint4*)(p);
  u.q[1] = *(const uint4*)(p + 16);
  return u.v;
}
__device__ __forceinline__ v8i ldsA_u8(const uint8_t* lds, int rowbase, int ld, int lane) {
  const uint8_t* p = lds + (rowbase + (lane & 15)) * ld + ((lane >> 4) << 4);
  union { v8i v; uint32_t u[8]; } x;
  x.u[0] = *(const uint32_t*)(p + 0);
  x.u[1] = *(const uint32_t*)(p + 4);
  x.u[2] = *(const uint32_t*)(p + 8);
  x.u[3] = *(const uint32_t*)(p + 12);
  x.u[4] = *(const uint32_t*)(p + 32);
  x.u[5] = *(const uint32_t*)(p + 36);
  x.u[6] = *(const uint32_t*)(p + 40);
  x.u[7] = *(const uint32_t*)(p + 44);
  return x.v;
}
__device__ __forceinline__ v8i ldsB_u8(const uint8_t* ldsT, int colbase, int ld, int lane) {
  const uint8_t* p = ldsT + (colbase + (lane & 15)) * ld + ((lane >> 4) << 4);
  union { v8i v; uint4 q[2]; } x;
  x.q[0] = *(const uint4*)(p);
  x.q[1] = *(const uint4*)(p + 32);
  return x.v;
}

__global__ __launch_bounds__(256) void cvt_f32_bf16_kernel(const float* __restrict__ in,
                                                           __bf16* __restrict__ out, long n) {
  long i = (long)blockIdx.x * 256 + threadIdx.x;
  if (i < n) out[i] = (__bf16)in[i];
}

__global__ __launch_bounds__(256) void cvt_w_kernel(const float* __restrict__ w,
                                                    const float* __restrict__ colscale,
                                                    __bf16* __restrict__ out) {
  int i = blockIdx.x * 256 + threadIdx.x;
  float s = colscale ? colscale[i & (D_ - 1)] : 1.0f;
  out[i] = (__bf16)(w[i] * s);
}

__global__ __launch_bounds__(512) void prep_affine_kernel(
    const float* gq, const float* bq, const float* mq, const float* vq,
    const float* gk, const float* bk, const float* mk, const float* vk,
    const float* gv, const float* bv, const float* mv, const float* vv,
    const float* gp, const float* bp, const float* mp, const float* vp,
    const float* wp, const float* gshift,
    float* aq, float* cq, float* ak, float* ck, float* av, float* cv,
    float* ap, float* cp, float* constp) {
  int p = threadIdx.x;
  if (p >= D_) return;
  float a;
  a = gq[p] * rsqrtf(vq[p] + 1e-5f); aq[p] = a; cq[p] = bq[p] - mq[p] * a;
  a = gk[p] * rsqrtf(vk[p] + 1e-5f); ak[p] = a; ck[p] = bk[p] - mk[p] * a;
  a = gv[p] * rsqrtf(vv[p] + 1e-5f); av[p] = a; cv[p] = bv[p] - mv[p] * a;
  a = gp[p] * rsqrtf(vp[p] + 1e-5f); ap[p] = a; cp[p] = bp[p] - mp[p] * a;
  float acc = 0.f;
  for (int d = 0; d < D_; ++d) acc += gshift[d] * wp[p * D_ + d];
  constp[p] = acc;
}

__global__ __launch_bounds__(256) void proj_lif_kernel(
    const __bf16* __restrict__ xbf, const __bf16* __restrict__ wbf,
    const float* __restrict__ affa, const float* __restrict__ affc,
    uint8_t* __restrict__ spk) {
  __shared__ __bf16 Xs[2][128 * 32];
  __shared__ __bf16 Ws[2][64 * 32];
  const int b = blockIdx.x, nb = blockIdx.y, db = blockIdx.z;
  const int tid = threadIdx.x, lane = tid & 31, wave = tid >> 5;
  const int wr = (wave & 3) * 32, wc = (wave >> 2) * 32;

  const int xe = tid * 16, xr = xe >> 5, xc = xe & 31;
  const int we = tid * 8, wrr = we >> 5, wcc = we & 31;
  const long wrow = (long)(db * 64 + wrr) * D_;

  v8f mem[2][2], acc[2][2];
#pragma unroll
  for (int i = 0; i < 2; ++i)
#pragma unroll
    for (int j = 0; j < 2; ++j) mem[i][j] = 0.0f;

  {
    const long rb0 = ((long)(0 * B_ + b)) * N_ + nb * 128;
    const __bf16* src = xbf + (rb0 + xr) * D_ + xc;
    async_ld_b128(&Xs[0][xr * 32 + xc], src);
    async_ld_b128(&Xs[0][xr * 32 + xc + 8], src + 8);
    async_ld_b128(&Ws[0][wrr * 32 + wcc], wbf + wrow + wcc);
  }

  const int TOT = T_ * (D_ / 32);
  for (int s = 0; s < TOT; ++s) {
    const int t = s >> 4, kk = s & 15;
    const long rowbase = ((long)(t * B_ + b)) * N_ + nb * 128;
    if (kk == 0) {
#pragma unroll
      for (int i = 0; i < 2; ++i)
#pragma unroll
        for (int j = 0; j < 2; ++j) acc[i][j] = 0.0f;
    }
    wait_async0();
    __syncthreads();
    if (s + 1 < TOT) {
      const int t2 = (s + 1) >> 4, k2 = ((s + 1) & 15) * 32;
      const long rb2 = ((long)(t2 * B_ + b)) * N_ + nb * 128;
      __bf16* Xn = Xs[(s + 1) & 1];
      __bf16* Wn = Ws[(s + 1) & 1];
      const __bf16* src = xbf + (rb2 + xr) * D_ + k2 + xc;
      async_ld_b128(&Xn[xr * 32 + xc], src);
      async_ld_b128(&Xn[xr * 32 + xc + 8], src + 8);
      async_ld_b128(&Wn[wrr * 32 + wcc], wbf + wrow + k2 + wcc);
    }
    const __bf16* Xc = Xs[s & 1];
    const __bf16* Wc = Ws[s & 1];
    v16bf a0 = ldsA_bf16(Xc, wr, 32, lane);
    v16bf a1 = ldsA_bf16(Xc, wr + 16, 32, lane);
    v16bf b0 = ldsB_bf16(Wc, wc, 32, lane);
    v16bf b1 = ldsB_bf16(Wc, wc + 16, 32, lane);
    acc[0][0] = wmma_bf16(a0, b0, acc[0][0]);
    acc[0][1] = wmma_bf16(a0, b1, acc[0][1]);
    acc[1][0] = wmma_bf16(a1, b0, acc[1][0]);
    acc[1][1] = wmma_bf16(a1, b1, acc[1][1]);

    if (kk == 15) {
#pragma unroll
      for (int i = 0; i < 2; ++i)
#pragma unroll
        for (int j = 0; j < 2; ++j) {
          int dcol = db * 64 + wc + j * 16 + (lane & 15);
          float ga = affa[dcol], gc = affc[dcol];
#pragma unroll
          for (int e = 0; e < 8; ++e) {
            int M = e + ((lane >> 4) << 3);
            float y = ga * acc[i][j][e] + gc;
            float m = mem[i][j][e];
            m += (y - m) * INV_TAU;
            float sv = (m >= VTH_) ? 1.f : 0.f;
            mem[i][j][e] = (sv != 0.f) ? 0.f : m;
            spk[(rowbase + wr + i * 16 + M) * D_ + dcol] = (uint8_t)sv;
          }
        }
    }
  }
}

__global__ __launch_bounds__(256) void attn_kernel(
    const uint8_t* __restrict__ qg, const uint8_t* __restrict__ kg,
    const uint8_t* __restrict__ vg, const float* __restrict__ btab,
    float* __restrict__ attn) {
  extern __shared__ uint8_t smem[];
  uint8_t* qs = smem;
  uint8_t* ks = smem + N_ * DH_;
  __bf16* vT = (__bf16*)(smem + 2 * N_ * DH_);
  __bf16* Ss = vT + DH_ * N_;

  const int tbh = blockIdx.x;
  const int h = tbh & (H_ - 1);
  const int tb = tbh >> 3;
  const long rowg = (long)tb * N_;
  const int tid = threadIdx.x, lane = tid & 31, wave = tid >> 5;

  for (int r = tid; r < N_; r += 256) {
    const uint8_t* s0 = qg + (rowg + r) * D_ + h * DH_;
    const uint8_t* s1 = kg + (rowg + r) * D_ + h * DH_;
#pragma unroll
    for (int c = 0; c < 4; ++c) {
      async_ld_b128(qs + r * DH_ + c * 16, s0 + c * 16);
      async_ld_b128(ks + r * DH_ + c * 16, s1 + c * 16);
    }
  }
  for (int e = tid; e < N_ * DH_; e += 256) {
    int n = e >> 6, d = e & 63;
    vT[d * N_ + n] = (__bf16)(float)vg[(rowg + n) * D_ + h * DH_ + d];
  }
  wait_async0();
  __syncthreads();

  __bf16* Sw = Ss + wave * 32 * 64;

  for (int nbk = 0; nbk < 2; ++nbk) {
    const int rowloc = wave * 64 + nbk * 32;
    v8i qa[2];
    qa[0] = ldsA_u8(qs, rowloc, DH_, lane);
    qa[1] = ldsA_u8(qs, rowloc + 16, DH_, lane);
    v8f acc[2][4];
#pragma unroll
    for (int i = 0; i < 2; ++i)
#pragma unroll
      for (int j = 0; j < 4; ++j) acc[i][j] = 0.0f;

    for (int mc = 0; mc < N_; mc += 64) {
      v8i kb[4];
#pragma unroll
      for (int jm = 0; jm < 4; ++jm) kb[jm] = ldsB_u8(ks, mc + jm * 16, DH_, lane);
#pragma unroll
      for (int i2 = 0; i2 < 2; ++i2) {
#pragma unroll
        for (int jm = 0; jm < 4; ++jm) {
          v8i z = {0, 0, 0, 0, 0, 0, 0, 0};
          v8i sc = wmma_iu8(qa[i2], kb[jm], z);
#pragma unroll
          for (int e = 0; e < 8; ++e) {
            int M = e + ((lane >> 4) << 3);
            int nl = rowloc + i2 * 16 + M;
            int ml = mc + jm * 16 + (lane & 15);
            float s = (float)sc[e] * SCALE_ + btab[(ml - nl + (N_ - 1)) * H_ + h];
            Sw[(i2 * 16 + M) * 64 + jm * 16 + (lane & 15)] = (__bf16)s;
          }
        }
      }
      asm volatile("s_wait_dscnt 0" ::: "memory");
#pragma unroll
      for (int ksb = 0; ksb < 2; ++ksb) {
        v16bf sa[2], vb[4];
        sa[0] = ldsA_bf16(Sw + ksb * 32, 0, 64, lane);
        sa[1] = ldsA_bf16(Sw + ksb * 32, 16, 64, lane);
#pragma unroll
        for (int j = 0; j < 4; ++j) vb[j] = ldsB_bf16(vT + mc + ksb * 32, j * 16, N_, lane);
#pragma unroll
        for (int i2 = 0; i2 < 2; ++i2)
#pragma unroll
          for (int j = 0; j < 4; ++j) acc[i2][j] = wmma_bf16(sa[i2], vb[j], acc[i2][j]);
      }
    }
#pragma unroll
    for (int i2 = 0; i2 < 2; ++i2)
#pragma unroll
      for (int j = 0; j < 4; ++j)
#pragma unroll
        for (int e = 0; e < 8; ++e) {
          int M = e + ((lane >> 4) << 3);
          int nl = rowloc + i2 * 16 + M;
          int dcol = h * DH_ + j * 16 + (lane & 15);
          attn[(rowg + nl) * D_ + dcol] = acc[i2][j][e];
        }
  }
}

__global__ __launch_bounds__(256) void lif_attn_kernel(const float* __restrict__ attn,
                                                       uint8_t* __restrict__ spk) {
  const long i = (long)blockIdx.x * 256 + threadIdx.x;
  const long stride = (long)B_ * N_ * D_;
  float m = 0.f;
#pragma unroll
  for (int t = 0; t < T_; ++t) {
    float y = attn[t * stride + i];
    m += (y - m) * INV_TAU;
    float s = (m >= VTH_) ? 1.f : 0.f;
    spk[t * stride + i] = (uint8_t)s;
    m = (s != 0.f) ? 0.f : m;
  }
}

__global__ __launch_bounds__(256) void proj_gate_lif_kernel(
    const uint8_t* __restrict__ sattn, const __bf16* __restrict__ wbf,
    const float* __restrict__ affa, const float* __restrict__ affc,
    const float* __restrict__ constp, float* __restrict__ out) {
  __shared__ uint8_t Xu[2][128 * 32];
  __shared__ __bf16 Ws[2][64 * 32];
  const int b = blockIdx.x, nb = blockIdx.y, db = blockIdx.z;
  const int tid = threadIdx.x, lane = tid & 31, wave = tid >> 5;
  const int wr = (wave & 3) * 32, wc = (wave >> 2) * 32;

  const int xe = tid * 16, xr = xe >> 5, xc = xe & 31;
  const int we = tid * 8, wrr = we >> 5, wcc = we & 31;
  const long wrow = (long)(db * 64 + wrr) * D_;

  v8f mem[2][2], acc[2][2];
#pragma unroll
  for (int i = 0; i < 2; ++i)
#pragma unroll
    for (int j = 0; j < 2; ++j) mem[i][j] = 0.0f;

  {
    const long rb0 = ((long)(0 * B_ + b)) * N_ + nb * 128;
    async_ld_b128(&Xu[0][xr * 32 + xc], sattn + (rb0 + xr) * D_ + xc);
    async_ld_b128(&Ws[0][wrr * 32 + wcc], wbf + wrow + wcc);
  }

  const int TOT = T_ * (D_ / 32);
  for (int s = 0; s < TOT; ++s) {
    const int t = s >> 4, kk = s & 15;
    const long rowbase = ((long)(t * B_ + b)) * N_ + nb * 128;
    if (kk == 0) {
#pragma unroll
      for (int i = 0; i < 2; ++i)
#pragma unroll
        for (int j = 0; j < 2; ++j) acc[i][j] = 0.0f;
    }
    wait_async0();
    __syncthreads();
    if (s + 1 < TOT) {
      const int t2 = (s + 1) >> 4, k2 = ((s + 1) & 15) * 32;
      const long rb2 = ((long)(t2 * B_ + b)) * N_ + nb * 128;
      async_ld_b128(&Xu[(s + 1) & 1][xr * 32 + xc], sattn + (rb2 + xr) * D_ + k2 + xc);
      async_ld_b128(&Ws[(s + 1) & 1][wrr * 32 + wcc], wbf + wrow + k2 + wcc);
    }
    const uint8_t* Xc = Xu[s & 1];
    const __bf16* Wc = Ws[s & 1];
    v16bf a0 = ldsA_u8bf(Xc, wr, 32, lane);
    v16bf a1 = ldsA_u8bf(Xc, wr + 16, 32, lane);
    v16bf b0 = ldsB_bf16(Wc, wc, 32, lane);
    v16bf b1 = ldsB_bf16(Wc, wc + 16, 32, lane);
    acc[0][0] = wmma_bf16(a0, b0, acc[0][0]);
    acc[0][1] = wmma_bf16(a0, b1, acc[0][1]);
    acc[1][0] = wmma_bf16(a1, b0, acc[1][0]);
    acc[1][1] = wmma_bf16(a1, b1, acc[1][1]);

    if (kk == 15) {
#pragma unroll
      for (int i = 0; i < 2; ++i)
#pragma unroll
        for (int j = 0; j < 2; ++j) {
          int dcol = db * 64 + wc + j * 16 + (lane & 15);
          float ga = affa[dcol], gc = affc[dcol], kp = constp[dcol];
#pragma unroll
          for (int e = 0; e < 8; ++e) {
            int M = e + ((lane >> 4) << 3);
            float y = ga * (kp + acc[i][j][e]) + gc;
            float m = mem[i][j][e];
            m += (y - m) * INV_TAU;
            float sv = (m >= VTH_) ? 1.f : 0.f;
            mem[i][j][e] = (sv != 0.f) ? 0.f : m;
            out[(rowbase + wr + i * 16 + M) * D_ + dcol] = sv;
          }
        }
    }
  }
}

extern "C" void kernel_launch(void* const* d_in, const int* in_sizes, int n_in,
                              void* d_out, int out_size, void* d_ws, size_t ws_size,
                              hipStream_t stream) {
  (void)in_sizes; (void)n_in; (void)out_size; (void)ws_size;
  const float* x   = (const float*)d_in[0];
  const float* Wq  = (const float*)d_in[1];
  const float* gq  = (const float*)d_in[2];
  const float* bq  = (const float*)d_in[3];
  const float* mq  = (const float*)d_in[4];
  const float* vq  = (const float*)d_in[5];
  const float* Wk  = (const float*)d_in[6];
  const float* gk  = (const float*)d_in[7];
  const float* bk  = (const float*)d_in[8];
  const float* mk  = (const float*)d_in[9];
  const float* vk  = (const float*)d_in[10];
  const float* Wv  = (const float*)d_in[11];
  const float* gv  = (const float*)d_in[12];
  const float* bv  = (const float*)d_in[13];
  const float* mv  = (const float*)d_in[14];
  const float* vv  = (const float*)d_in[15];
  const float* Wp  = (const float*)d_in[16];
  const float* gp  = (const float*)d_in[17];
  const float* bp  = (const float*)d_in[18];
  const float* mp  = (const float*)d_in[19];
  const float* vp  = (const float*)d_in[20];
  const float* btab = (const float*)d_in[21];
  const float* gsc  = (const float*)d_in[22];
  const float* gsh  = (const float*)d_in[23];
  float* out = (float*)d_out;

  char* ws = (char*)d_ws;
  size_t off = 0;
  auto take = [&](size_t bytes) -> char* {
    char* p = ws + off;
    off = (off + bytes + 255) & ~(size_t)255;
    return p;
  };
  __bf16* xbf = (__bf16*)take((size_t)TBN_ * D_ * 2);
  __bf16* wqb = (__bf16*)take((size_t)D_ * D_ * 2);
  __bf16* wkb = (__bf16*)take((size_t)D_ * D_ * 2);
  __bf16* wvb = (__bf16*)take((size_t)D_ * D_ * 2);
  __bf16* wpb = (__bf16*)take((size_t)D_ * D_ * 2);
  float* aq = (float*)take(D_ * 4); float* cq = (float*)take(D_ * 4);
  float* ak = (float*)take(D_ * 4); float* ck = (float*)take(D_ * 4);
  float* av = (float*)take(D_ * 4); float* cv = (float*)take(D_ * 4);
  float* ap = (float*)take(D_ * 4); float* cp = (float*)take(D_ * 4);
  float* constp = (float*)take(D_ * 4);
  uint8_t* sq = (uint8_t*)take((size_t)TBN_ * D_);
  uint8_t* sk = (uint8_t*)take((size_t)TBN_ * D_);
  uint8_t* sv = (uint8_t*)take((size_t)TBN_ * D_);
  float* attn = (float*)take((size_t)TBN_ * D_ * 4);
  uint8_t* sat = (uint8_t*)take((size_t)TBN_ * D_);

  const long nx = (long)TBN_ * D_;
  cvt_f32_bf16_kernel<<<(int)((nx + 255) / 256), 256, 0, stream>>>(x, xbf, nx);
  cvt_w_kernel<<<(D_ * D_) / 256, 256, 0, stream>>>(Wq, nullptr, wqb);
  cvt_w_kernel<<<(D_ * D_) / 256, 256, 0, stream>>>(Wk, nullptr, wkb);
  cvt_w_kernel<<<(D_ * D_) / 256, 256, 0, stream>>>(Wv, nullptr, wvb);
  cvt_w_kernel<<<(D_ * D_) / 256, 256, 0, stream>>>(Wp, gsc, wpb);
  prep_affine_kernel<<<1, 512, 0, stream>>>(gq, bq, mq, vq, gk, bk, mk, vk,
                                            gv, bv, mv, vv, gp, bp, mp, vp,
                                            Wp, gsh, aq, cq, ak, ck, av, cv, ap, cp,
                                            constp);

  dim3 g1(B_, N_ / 128, D_ / 64);
  proj_lif_kernel<<<g1, 256, 0, stream>>>(xbf, wqb, aq, cq, sq);
  proj_lif_kernel<<<g1, 256, 0, stream>>>(xbf, wkb, ak, ck, sk);
  proj_lif_kernel<<<g1, 256, 0, stream>>>(xbf, wvb, av, cv, sv);

  const size_t smem = (size_t)2 * N_ * DH_
                    + (size_t)DH_ * N_ * 2
                    + (size_t)8 * 32 * 64 * 2;
  attn_kernel<<<T_ * B_ * H_, 256, smem, stream>>>(sq, sk, sv, btab, attn);

  lif_attn_kernel<<<(B_ * N_ * D_) / 256, 256, 0, stream>>>(attn, sat);
  proj_gate_lif_kernel<<<g1, 256, 0, stream>>>(sat, wpb, ap, cp, constp, out);
}
